// MLA_9294309228727
// MI455X (gfx1250) — matrix operand order repaired and checked by reading, not compiled, not run on hardware
//
#include <hip/hip_runtime.h>
#include <hip/hip_bf16.h>
#include <math.h>

// ---------------- problem constants ----------------
#define BATCH 2
#define TSEQ  2048
#define MODEL 2048
#define NH    16
#define LAT   512
#define DKC   128        // content dim per head
#define DHRR  64         // rope dim per head
#define DQK   192        // DKC + DHRR
#define QBLK  128        // query rows per workgroup in flash
#define KBLK  32         // key block (one WMMA K-step)

typedef __bf16 bf16;
typedef __attribute__((ext_vector_type(16))) __bf16 bf16x16;
typedef __attribute__((ext_vector_type(8)))  __bf16 bf16x8;
typedef __attribute__((ext_vector_type(8)))  float  f32x8;
typedef __attribute__((ext_vector_type(4)))  unsigned int u32x4;
typedef __attribute__((ext_vector_type(4)))  int i32x4;
typedef __attribute__((ext_vector_type(8)))  int i32x8;

#if defined(__has_builtin)
#  if __has_builtin(__builtin_amdgcn_tensor_load_to_lds)
#    define HAVE_TDM 1
#  endif
#endif
#ifndef HAVE_TDM
#  define HAVE_TDM 0
#endif

// ---------------- WMMA fragment helpers ----------------
__device__ __forceinline__ bf16x16 frag_cat(bf16x8 lo, bf16x8 hi) {
  bf16x16 r;
#pragma unroll
  for (int i = 0; i < 8; ++i) { r[i] = lo[i]; r[i + 8] = hi[i]; }
  return r;
}

// A fragment (16x32 bf16, row-major source): lane holds row m=lane&15.
// ISA layout: VGPR0-3 k = half*8 + 0..7 ; VGPR4-7 k = 16 + half*8 + 0..7.
__device__ __forceinline__ bf16x16 load_frag_a(const bf16* p, int lda, int lane) {
  const int m = lane & 15, half = lane >> 4;
  const bf16* r = p + (size_t)m * lda + half * 8;
  bf16x8 lo = *(const bf16x8*)(r);
  bf16x8 hi = *(const bf16x8*)(r + 16);
  return frag_cat(lo, hi);
}

// B fragment (32x16 bf16) from an N-major operand (ldb strides over k):
// lane holds column n=lane&15, k = half*8 + 0..7 and 16 + half*8 + 0..7.
__device__ __forceinline__ bf16x16 load_frag_b(const bf16* p, int ldb, int lane) {
  const int n = lane & 15, half = lane >> 4;
  const bf16* r = p + (size_t)n * ldb + half * 8;
  bf16x8 lo = *(const bf16x8*)(r);
  bf16x8 hi = *(const bf16x8*)(r + 16);
  return frag_cat(lo, hi);
}

__device__ __forceinline__ f32x8 wmma_bf16f32(bf16x16 a, bf16x16 b, f32x8 c) {
  return __builtin_amdgcn_wmma_f32_16x16x32_bf16(false, a, false, b, (short)0, c,
                                                 false, false);
}

// ---------------- Tensor Data Mover (CDNA5 TDM) ----------------
#if HAVE_TDM
// Issue one TDM 2-D tile load (ISA 08_async_tensor §8 descriptor layout).
// Dims/strides in data_size(=2B) units. tile_d1==0 -> single row (1-D).
__device__ __forceinline__ void tdm_load_2d(unsigned int ldsAddr, const void* gptr,
                                            unsigned int tensor_d0, unsigned int tensor_d1,
                                            unsigned int tile_d0, unsigned int tile_d1,
                                            unsigned long long stride0) {
  const unsigned long long ga = (unsigned long long)(uintptr_t)gptr;
  u32x4 g0;
  g0[0] = 1u;                                            // count=1, user D#
  g0[1] = ldsAddr;                                       // LDS byte address
  g0[2] = (unsigned int)ga;                              // global_addr[31:0]
  g0[3] = (unsigned int)((ga >> 32) & 0x01FFFFFFu) | 0x80000000u;  // addr[56:32] | type=2
  i32x8 g1;
  g1[0] = (int)(1u << 16);                               // wg_mask=0, data_size=1 (2B)
  g1[1] = (int)((tensor_d0 & 0xFFFFu) << 16);            // tensor_dim0[15:0] @ bits 63:48
  g1[2] = (int)((tensor_d0 >> 16) | ((tensor_d1 & 0xFFFFu) << 16));
  g1[3] = (int)((tensor_d1 >> 16) | (tile_d0 << 16));    // tile_dim0 @ bits 127:112
  g1[4] = (int)(tile_d1);                                // tile_dim1, tile_dim2=0
  g1[5] = (int)(unsigned int)stride0;                    // tensor_dim0_stride[31:0]
  g1[6] = (int)(unsigned int)((stride0 >> 32) & 0xFFFFu);// stride[47:32], dim1_stride=0
  g1[7] = 0;
  const i32x4 z4 = {0, 0, 0, 0};
#if __clang_major__ >= 23
  const i32x8 z8 = {0, 0, 0, 0, 0, 0, 0, 0};
  __builtin_amdgcn_tensor_load_to_lds(g0, g1, z4, z4, z8, 0);
#else
  __builtin_amdgcn_tensor_load_to_lds(g0, g1, z4, z4, 0);
#endif
}

__device__ __forceinline__ void tdm_wait0() {
#if __has_builtin(__builtin_amdgcn_s_wait_tensorcnt)
  __builtin_amdgcn_s_wait_tensorcnt(0);
#else
  asm volatile("s_wait_tensorcnt 0" ::: "memory");
#endif
}
#endif  // HAVE_TDM

// ---------------- pack kernels ----------------
__global__ __launch_bounds__(256) void cvt_f32_bf16_kernel(const float* __restrict__ in,
                                                           bf16* __restrict__ out, int n) {
  int i = blockIdx.x * 256 + threadIdx.x;
  if (i < n) out[i] = (bf16)in[i];
}

// W[K,N] f32 row-major -> Wt[N,K] bf16 (N-major for B fragments)
__global__ __launch_bounds__(256) void transpose_cvt_kernel(const float* __restrict__ W,
                                                            bf16* __restrict__ Wt,
                                                            int K, int N) {
  __shared__ float tile[32][33];
  const int k0 = blockIdx.y * 32, n0 = blockIdx.x * 32;
  const int tx = threadIdx.x & 31, ty = threadIdx.x >> 5;
#pragma unroll
  for (int i = 0; i < 32; i += 8) {
    int k = k0 + ty + i, n = n0 + tx;
    tile[ty + i][tx] = (k < K && n < N) ? W[(size_t)k * N + n] : 0.f;
  }
  __syncthreads();
#pragma unroll
  for (int i = 0; i < 32; i += 8) {
    int n = n0 + ty + i, k = k0 + tx;
    if (n < N && k < K) Wt[(size_t)n * K + k] = (bf16)tile[tx][ty + i];
  }
}

// ---------------- generic WMMA GEMM ----------------
// C[M,N] = A[M,K] (row-major bf16) x Bt[N,K] (N-major bf16)
// MODE 0: bf16 row-major store (ldc)
// MODE 1: bf16 scatter into [B,NH,T,DQK] at d<128 (Q/K content part)
// MODE 2: bf16 scatter into [B,NH,DKC,T]   (V transposed for PV)
// MODE 3: f32 row-major store (final output)
template <int MODE>
__global__ __launch_bounds__(256) void gemm_bf16_kernel(const bf16* __restrict__ A,
                                                        const bf16* __restrict__ Bt,
                                                        void* __restrict__ Cout,
                                                        int M, int N, int K,
                                                        int lda, int ldb, int ldc) {
  const int tid = threadIdx.x, wave = tid >> 5, lane = tid & 31;
  const int mt = wave >> 2, nt = wave & 3;
  const int m0 = (blockIdx.y * 2 + mt) * 16;
  const int n0 = (blockIdx.x * 4 + nt) * 16;
  if (m0 >= M || n0 >= N) return;   // wave-uniform; EXEC stays all-ones for WMMA

  f32x8 acc = {0.f, 0.f, 0.f, 0.f, 0.f, 0.f, 0.f, 0.f};
  const bf16* ap = A + (size_t)m0 * lda;
  const bf16* bp = Bt + (size_t)n0 * ldb;
#pragma unroll 4
  for (int kk = 0; kk < K; kk += 32) {
    if (kk + 128 < K) {             // pull next k-tiles into GL2 (global_prefetch_b8)
      __builtin_prefetch(ap + kk + 128, 0, 1);
      __builtin_prefetch(bp + kk + 128, 0, 1);
    }
    bf16x16 a = load_frag_a(ap + kk, lda, lane);
    bf16x16 b = load_frag_b(bp + kk, ldb, lane);
    acc = wmma_bf16f32(a, b, acc);
  }

  const int half = lane >> 4, nn = lane & 15;
#pragma unroll
  for (int r = 0; r < 8; ++r) {
    const int m = m0 + half * 8 + r;
    const int n = n0 + nn;
    const float v = acc[r];
    if (MODE == 0) {
      ((bf16*)Cout)[(size_t)m * ldc + n] = (bf16)v;
    } else if (MODE == 1) {
      int b = m >> 11, t = m & (TSEQ - 1);
      int h = n >> 7, d = n & (DKC - 1);
      ((bf16*)Cout)[(size_t)((b * NH + h) * TSEQ + t) * DQK + d] = (bf16)v;
    } else if (MODE == 2) {
      int b = m >> 11, t = m & (TSEQ - 1);
      int h = n >> 7, d = n & (DKC - 1);
      ((bf16*)Cout)[(size_t)((b * NH + h) * DKC + d) * TSEQ + t] = (bf16)v;
    } else {
      ((float*)Cout)[(size_t)m * ldc + n] = v;
    }
  }
}

// ---------------- RoPE kernels (faithful to reference tables) ----------------
// theta_j = 10000 * (-2*(floor(j/2)-1)/D); pair-swap is over the TIME axis:
// a[g] = x[g^1]  (from reshape(-1,2,D)[:, ::-1, :])
__global__ __launch_bounds__(256) void rope_kr_kernel(const bf16* __restrict__ xkr,
                                                      bf16* __restrict__ Kbuf) {
  int idx = blockIdx.x * 256 + threadIdx.x;          // over (B*T)*64
  if (idx >= BATCH * TSEQ * DHRR) return;
  const int g = idx >> 6, j = idx & 63;
  const int b = g >> 11, t = g & (TSEQ - 1);
  const float pos = (float)(j >> 1);
  const float theta = 10000.0f * (-2.0f * (pos - 1.0f) / (float)DHRR);
  const float ang = (float)t * theta;
  const float v = (float)xkr[idx] * cosf(ang) +
                  (float)xkr[(size_t)(g ^ 1) * DHRR + j] * sinf(ang);
  const bf16 vb = (bf16)v;
#pragma unroll
  for (int h = 0; h < NH; ++h)       // shared rope-key broadcast to every head
    Kbuf[(size_t)((b * NH + h) * TSEQ + t) * DQK + DKC + j] = vb;
}

__global__ __launch_bounds__(256) void rope_qr_kernel(const bf16* __restrict__ qr,
                                                      bf16* __restrict__ Qbuf) {
  int idx = blockIdx.x * 256 + threadIdx.x;          // over (B*T)*1024
  if (idx >= BATCH * TSEQ * NH * DHRR) return;
  const int g = idx >> 10, j = idx & 1023;
  const int b = g >> 11, t = g & (TSEQ - 1);
  const float pos = (float)(j >> 1);
  const float theta = 10000.0f * (-2.0f * (pos - 1.0f) / (float)(NH * DHRR));
  const float ang = (float)t * theta;
  const float v = (float)qr[idx] * cosf(ang) +
                  (float)qr[(size_t)(g ^ 1) * (NH * DHRR) + j] * sinf(ang);
  const int h = j >> 6, d = j & 63;
  Qbuf[(size_t)((b * NH + h) * TSEQ + t) * DQK + DKC + d] = (bf16)v;
}

// ---------------- causal flash attention ----------------
// Q,K: [B,NH,T,DQK] bf16 ; Vt: [B,NH,DKC,T] bf16 ; O: [B,T,MODEL] bf16
__global__ __launch_bounds__(256) void flash_kernel(const bf16* __restrict__ Q,
                                                    const bf16* __restrict__ Kb,
                                                    const bf16* __restrict__ Vt,
                                                    bf16* __restrict__ O) {
  __shared__ __align__(16) bf16 Ks[KBLK][DQK];       // 12 KB
  __shared__ __align__(16) bf16 Vs[DKC][KBLK];       //  8 KB
  __shared__ __align__(16) bf16 Ps[8][16][KBLK];     //  8 KB per-wave P staging

  const int tid = threadIdx.x, wave = tid >> 5, lane = tid & 31;
  const int qb = blockIdx.x, h = blockIdx.y, b = blockIdx.z;
  const int half = lane >> 4, nn = lane & 15;
  const int qrow0 = qb * QBLK + wave * 16;

  const bf16* Qhead = Q  + (size_t)((b * NH + h) * TSEQ) * DQK;
  const bf16* Khead = Kb + (size_t)((b * NH + h) * TSEQ) * DQK;
  const bf16* Vhead = Vt + (size_t)((b * NH + h) * DKC) * TSEQ;

  // 16x192 Q tile as six 16x32 A fragments, kept in VGPRs for the whole pass
  bf16x16 aq[6];
#pragma unroll
  for (int c = 0; c < 6; ++c)
    aq[c] = load_frag_a(Qhead + (size_t)qrow0 * DQK + c * 32, DQK, lane);

  const f32x8 zero8 = {0.f, 0.f, 0.f, 0.f, 0.f, 0.f, 0.f, 0.f};
  f32x8 acc[8];
#pragma unroll
  for (int dt = 0; dt < 8; ++dt) acc[dt] = zero8;
  float mrow[8], lrow[8];
#pragma unroll
  for (int r = 0; r < 8; ++r) { mrow[r] = -1e30f; lrow[r] = 0.f; }

  const float SCL2 = 0.12751744f;        // (1/sqrt(128)) * log2(e)
  const int kend = qb * QBLK + QBLK;     // causal bound, uniform per block

#if HAVE_TDM
  const unsigned int ksLds = (unsigned int)(uintptr_t)(&Ks[0][0]);
  const unsigned int vsLds = (unsigned int)(uintptr_t)(&Vs[0][0]);
#endif

  for (int kb = 0; kb < kend; kb += KBLK) {
    __syncthreads();                     // previous block fully consumed
#if HAVE_TDM
    if (wave == 0) {
      // K block: 32 rows x 192 = 12 KB contiguous -> 1-D tile
      tdm_load_2d(ksLds, Khead + (size_t)kb * DQK,
                  KBLK * DQK, 1, KBLK * DQK, 0, KBLK * DQK);
      // V^T block: 128 rows x 32 keys, row stride T
      tdm_load_2d(vsLds, Vhead + kb, TSEQ, DKC, KBLK, DKC, TSEQ);
      tdm_wait0();                       // TENSORcnt==0: DMA landed in LDS
    }
#else
    {
      const bf16x8* src = (const bf16x8*)(Khead + (size_t)kb * DQK);
      bf16x8* dst = (bf16x8*)(&Ks[0][0]);
      for (int i = tid; i < KBLK * DQK / 8; i += 256) dst[i] = src[i];
      bf16x8* vdst = (bf16x8*)(&Vs[0][0]);
      for (int i = tid; i < DKC * KBLK / 8; i += 256) {
        int d = i >> 2, jv = i & 3;
        vdst[i] = ((const bf16x8*)(Vhead + (size_t)d * TSEQ + kb))[jv];
      }
    }
#endif
    __syncthreads();

    // S = Q K^T over two 16-key tiles, scaled into log2 domain + causal mask
    float s2[2][8];
#pragma unroll
    for (int t = 0; t < 2; ++t) {
      f32x8 s = zero8;
#pragma unroll
      for (int c = 0; c < 6; ++c)
        s = wmma_bf16f32(aq[c], load_frag_b(&Ks[t * 16][c * 32], DQK, lane), s);
      const int key = kb + t * 16 + nn;
#pragma unroll
      for (int r = 0; r < 8; ++r) {
        const int qrow = qrow0 + half * 8 + r;
        s2[t][r] = (key <= qrow) ? s[r] * SCL2 : -1e30f;
      }
    }

    // online softmax; row reductions across the 16 lanes of each half (wave32)
#pragma unroll
    for (int r = 0; r < 8; ++r) {
      float rm = fmaxf(s2[0][r], s2[1][r]);
      rm = fmaxf(rm, __shfl_xor(rm, 1, 32));
      rm = fmaxf(rm, __shfl_xor(rm, 2, 32));
      rm = fmaxf(rm, __shfl_xor(rm, 4, 32));
      rm = fmaxf(rm, __shfl_xor(rm, 8, 32));
      const float mnew = fmaxf(mrow[r], rm);
      const float f = exp2f(mrow[r] - mnew);
      mrow[r] = mnew;
      const float p0 = exp2f(s2[0][r] - mnew);
      const float p1 = exp2f(s2[1][r] - mnew);
      float rs = p0 + p1;
      rs += __shfl_xor(rs, 1, 32);
      rs += __shfl_xor(rs, 2, 32);
      rs += __shfl_xor(rs, 4, 32);
      rs += __shfl_xor(rs, 8, 32);
      lrow[r] = lrow[r] * f + rs;
#pragma unroll
      for (int dt = 0; dt < 8; ++dt) acc[dt][r] *= f;
      // D-layout -> row-major LDS so P can be re-read as an A fragment
      Ps[wave][half * 8 + r][nn]      = (bf16)p0;
      Ps[wave][half * 8 + r][16 + nn] = (bf16)p1;
    }

    asm volatile("s_wait_dscnt 0" ::: "memory");   // wave-local LDS RAW fence
    bf16x16 ap = load_frag_a(&Ps[wave][0][0], KBLK, lane);
#pragma unroll
    for (int dt = 0; dt < 8; ++dt)
      acc[dt] = wmma_bf16f32(ap, load_frag_b(&Vs[dt * 16][0], KBLK, lane), acc[dt]);
  }

  // epilogue: normalize and scatter heads back to [B,T,MODEL]
#pragma unroll
  for (int dt = 0; dt < 8; ++dt) {
#pragma unroll
    for (int r = 0; r < 8; ++r) {
      const int t = qrow0 + half * 8 + r;
      const float v = acc[dt][r] / lrow[r];
      O[(size_t)(b * TSEQ + t) * MODEL + h * DKC + dt * 16 + nn] = (bf16)v;
    }
  }
}

// ---------------- host orchestration ----------------
extern "C" void kernel_launch(void* const* d_in, const int* in_sizes, int n_in,
                              void* d_out, int out_size, void* d_ws, size_t ws_size,
                              hipStream_t stream) {
  (void)in_sizes; (void)n_in; (void)out_size; (void)ws_size;
  const float* x    = (const float*)d_in[0];
  const float* Wdkv = (const float*)d_in[1];
  const float* Wuk  = (const float*)d_in[2];
  const float* Wuv  = (const float*)d_in[3];
  const float* Wdq  = (const float*)d_in[4];
  const float* Wuq  = (const float*)d_in[5];
  const float* Wkr  = (const float*)d_in[6];
  const float* Wqr  = (const float*)d_in[7];
  const float* Wo   = (const float*)d_in[8];
  float* out = (float*)d_out;

  char* w = (char*)d_ws;
  auto alloc = [&](size_t elems) {
    bf16* p = (bf16*)w;
    w += ((elems * sizeof(bf16) + 255) & ~(size_t)255);
    return p;
  };

  const size_t MT = (size_t)BATCH * TSEQ;            // 4096 rows
  bf16* xb    = alloc(MT * MODEL);
  bf16* WdkvT = alloc((size_t)LAT * MODEL);
  bf16* WdqT  = alloc((size_t)LAT * MODEL);
  bf16* WkrT  = alloc((size_t)DHRR * MODEL);
  bf16* WqrT  = alloc((size_t)(NH * DHRR) * LAT);
  bf16* WukT  = alloc((size_t)MODEL * LAT);
  bf16* WuvT  = alloc((size_t)MODEL * LAT);
  bf16* WuqT  = alloc((size_t)MODEL * LAT);
  bf16* WoT   = alloc((size_t)MODEL * MODEL);
  bf16* cKV   = alloc(MT * LAT);
  bf16* cq    = alloc(MT * LAT);
  bf16* xkr   = alloc(MT * DHRR);
  bf16* qrraw = alloc(MT * (NH * DHRR));
  bf16* Qbuf  = alloc((size_t)BATCH * NH * TSEQ * DQK);
  bf16* Kbuf  = alloc((size_t)BATCH * NH * TSEQ * DQK);
  bf16* Vtb   = alloc((size_t)BATCH * NH * DKC * TSEQ);
  bf16* Obuf  = alloc(MT * MODEL);

  const dim3 tb(256);
  // pack x
  cvt_f32_bf16_kernel<<<(int)((MT * MODEL + 255) / 256), tb, 0, stream>>>(x, xb,
                                                               (int)(MT * MODEL));
  // pack weights (transpose to N-major bf16)
  auto tgrid = [](int K, int N) { return dim3((N + 31) / 32, (K + 31) / 32); };
  transpose_cvt_kernel<<<tgrid(MODEL, LAT),   tb, 0, stream>>>(Wdkv, WdkvT, MODEL, LAT);
  transpose_cvt_kernel<<<tgrid(MODEL, LAT),   tb, 0, stream>>>(Wdq,  WdqT,  MODEL, LAT);
  transpose_cvt_kernel<<<tgrid(MODEL, DHRR),  tb, 0, stream>>>(Wkr,  WkrT,  MODEL, DHRR);
  transpose_cvt_kernel<<<tgrid(LAT, NH*DHRR), tb, 0, stream>>>(Wqr,  WqrT,  LAT, NH * DHRR);
  transpose_cvt_kernel<<<tgrid(LAT, MODEL),   tb, 0, stream>>>(Wuk,  WukT,  LAT, MODEL);
  transpose_cvt_kernel<<<tgrid(LAT, MODEL),   tb, 0, stream>>>(Wuv,  WuvT,  LAT, MODEL);
  transpose_cvt_kernel<<<tgrid(LAT, MODEL),   tb, 0, stream>>>(Wuq,  WuqT,  LAT, MODEL);
  transpose_cvt_kernel<<<tgrid(MODEL, MODEL), tb, 0, stream>>>(Wo,   WoT,   MODEL, MODEL);

  auto ggrid = [](int M, int N) { return dim3(N / 64, M / 32); };
  // down-projections + rope raw parts
  gemm_bf16_kernel<0><<<ggrid(4096, LAT),  tb, 0, stream>>>(xb, WdkvT, cKV,   4096, LAT, MODEL, MODEL, MODEL, LAT);
  gemm_bf16_kernel<0><<<ggrid(4096, LAT),  tb, 0, stream>>>(xb, WdqT,  cq,    4096, LAT, MODEL, MODEL, MODEL, LAT);
  gemm_bf16_kernel<0><<<ggrid(4096, DHRR), tb, 0, stream>>>(xb, WkrT,  xkr,   4096, DHRR, MODEL, MODEL, MODEL, DHRR);
  gemm_bf16_kernel<0><<<ggrid(4096, NH*DHRR), tb, 0, stream>>>(cq, WqrT, qrraw, 4096, NH * DHRR, LAT, LAT, LAT, NH * DHRR);
  // up-projections scattered into attention layouts
  gemm_bf16_kernel<1><<<ggrid(4096, MODEL), tb, 0, stream>>>(cKV, WukT, Kbuf, 4096, MODEL, LAT, LAT, LAT, 0);
  gemm_bf16_kernel<1><<<ggrid(4096, MODEL), tb, 0, stream>>>(cq,  WuqT, Qbuf, 4096, MODEL, LAT, LAT, LAT, 0);
  gemm_bf16_kernel<2><<<ggrid(4096, MODEL), tb, 0, stream>>>(cKV, WuvT, Vtb,  4096, MODEL, LAT, LAT, LAT, 0);
  // rope fills Q/K[..., 128:192]
  rope_kr_kernel<<<(BATCH * TSEQ * DHRR) / 256,      tb, 0, stream>>>(xkr,   Kbuf);
  rope_qr_kernel<<<(BATCH * TSEQ * NH * DHRR) / 256, tb, 0, stream>>>(qrraw, Qbuf);
  // causal flash attention
  flash_kernel<<<dim3(TSEQ / QBLK, NH, BATCH), tb, 0, stream>>>(Qbuf, Kbuf, Vtb, Obuf);
  // output projection, f32 store
  gemm_bf16_kernel<3><<<ggrid(4096, MODEL), tb, 0, stream>>>(Obuf, WoT, out, 4096, MODEL, MODEL, MODEL, MODEL, MODEL);
}
